// S_ChnAttention2d_36919538876801
// MI455X (gfx1250) — matrix operand order repaired and checked by reading, not compiled, not run on hardware
//
#include <hip/hip_runtime.h>
#include <hip/hip_bf16.h>
#include <math.h>

// ---------------------------------------------------------------------------
// Channel self-attention (N=16, C=128, H*W=16384) for MI455X / gfx1250.
//
//   G = X X^T (per batch, K=16384)       -> bf16 WMMA, fp32 atomically-accumulated
//   W = softmax_rows(G)                  -> tiny
//   M = conv_w @ W                       -> tiny VALU matmul (algebraic fusion:
//                                           conv_w @ (W @ X) == (conv_w @ W) @ X)
//   out = M @ X + b                      -> bf16 WMMA, bias fused
//
// Roofline: ~400 MB HBM traffic (~17us @ 23.3 TB/s) vs ~17 GFLOP of bf16 WMMA
// (few us) -> memory bound, so bf16-in/f32-acc is the right precision/path.
// Workspace: G (1 MB) + M (1 MB) = 2 MB of d_ws.
// ---------------------------------------------------------------------------

typedef __attribute__((ext_vector_type(16))) __bf16           v16bf;
typedef __attribute__((ext_vector_type(8)))  float            v8f;
typedef __attribute__((ext_vector_type(8)))  unsigned short   v8us;
typedef __attribute__((ext_vector_type(16))) unsigned short   v16us;

#define NB      16
#define CC      128
#define LL      16384
#define KCHUNK  1024

// fp32 -> bf16 round-to-nearest-even
static __device__ __forceinline__ unsigned short f2bf(float f) {
    unsigned u = __builtin_bit_cast(unsigned, f);
    u += 0x7FFFu + ((u >> 16) & 1u);
    return (unsigned short)(u >> 16);
}

// Assemble a 16x32 A-frag / 32x16 B-frag (ISA 7.12.2 16-bit layouts) from two
// 16-byte aligned runs of 8 bf16 each.
static __device__ __forceinline__ v16bf make_frag(const unsigned short* p0,
                                                  const unsigned short* p1) {
    v8us lo = *(const v8us*)p0;
    v8us hi = *(const v8us*)p1;
    v16us w = __builtin_shufflevector(lo, hi, 0,1,2,3,4,5,6,7,8,9,10,11,12,13,14,15);
    return __builtin_bit_cast(v16bf, w);
}

// A-fragment straight from global fp32: two contiguous runs of 8 floats,
// 16 floats apart (K layout: {k..k+7} and {k+16..k+23} for this lane half).
static __device__ __forceinline__ v16bf frag_f32(const float* p0) {
    float4 a0 = *(const float4*)(p0);
    float4 a1 = *(const float4*)(p0 + 4);
    float4 b0 = *(const float4*)(p0 + 16);
    float4 b1 = *(const float4*)(p0 + 20);
    v16us w;
    w[0]=f2bf(a0.x);  w[1]=f2bf(a0.y);  w[2]=f2bf(a0.z);  w[3]=f2bf(a0.w);
    w[4]=f2bf(a1.x);  w[5]=f2bf(a1.y);  w[6]=f2bf(a1.z);  w[7]=f2bf(a1.w);
    w[8]=f2bf(b0.x);  w[9]=f2bf(b0.y);  w[10]=f2bf(b0.z); w[11]=f2bf(b0.w);
    w[12]=f2bf(b1.x); w[13]=f2bf(b1.y); w[14]=f2bf(b1.z); w[15]=f2bf(b1.w);
    return __builtin_bit_cast(v16bf, w);
}

// ---------------------------------------------------------------------------
// Kernel 1: partial Gram  G[n] += X[n][:, k0:k0+1024] @ X[n][:, k0:k0+1024]^T
// grid = (16 k-chunks, 16 batches), 256 threads = 8 waves.
// ---------------------------------------------------------------------------
__global__ __launch_bounds__(256) void gram_kernel(const float* __restrict__ x,
                                                   float* __restrict__ G) {
    // 128 rows x 32 k as bf16, row stride 40 shorts (80B): 16B aligned frags,
    // conflict-free bank pattern (20 DW stride).
    __shared__ __attribute__((aligned(16))) unsigned short lds[CC * 40];

    const int chunk = blockIdx.x;
    const int n     = blockIdx.y;
    const int t     = threadIdx.x;
    const int wave  = t >> 5;
    const int lane  = t & 31;
    const int half  = lane >> 4;
    const int ln    = lane & 15;
    const float* xb = x + (size_t)n * CC * LL;
    const int k0    = chunk * KCHUNK;

    v8f acc[8] = {};

    for (int ks = 0; ks < KCHUNK / 32; ++ks) {
        const int kbase = k0 + ks * 32;
        // stage 128x32 fp32 -> bf16 LDS (coalesced float4 reads)
        for (int i = 0; i < 4; ++i) {
            int f   = t + 256 * i;        // float4 index 0..1023
            int row = f >> 3;             // 8 float4 per row
            int fk  = (f & 7) * 4;
            const float4 v = *(const float4*)(xb + (size_t)row * LL + kbase + fk);
            unsigned short* d = &lds[row * 40 + fk];
            d[0] = f2bf(v.x); d[1] = f2bf(v.y); d[2] = f2bf(v.z); d[3] = f2bf(v.w);
        }
        __syncthreads();

        const int m = 16 * wave + ln;     // this wave's 16-row stripe of A
        v16bf a = make_frag(&lds[m * 40 + half * 8],
                            &lds[m * 40 + 16 + half * 8]);
        for (int j = 0; j < 8; ++j) {     // sweep 8 column tiles (B = X^T rows)
            const int r = 16 * j + ln;
            v16bf b = make_frag(&lds[r * 40 + half * 8],
                                &lds[r * 40 + 16 + half * 8]);
            acc[j] = __builtin_amdgcn_wmma_f32_16x16x32_bf16(
                         false, a, false, b, (short)0, acc[j], false, false);
        }
        __syncthreads();
    }

    float* g = G + (size_t)n * CC * CC;
    for (int j = 0; j < 8; ++j) {
        const int col = 16 * j + ln;
        for (int v = 0; v < 8; ++v) {
            const int row = 16 * wave + v + 8 * half;   // C/D layout (ISA 7.12.2)
            atomicAdd(&g[row * CC + col], acc[j][v]);
        }
    }
}

// ---------------------------------------------------------------------------
// Kernel 2: in-place row softmax of G. grid = 16 batches, 128 threads (1/row).
// ---------------------------------------------------------------------------
__global__ __launch_bounds__(128) void softmax_kernel(float* __restrict__ G) {
    const int n = blockIdx.x;
    const int c = threadIdx.x;
    float* row = G + ((size_t)n * CC + c) * CC;
    float m = -INFINITY;
    for (int d = 0; d < CC; ++d) m = fmaxf(m, row[d]);
    float s = 0.f;
    for (int d = 0; d < CC; ++d) { float e = __expf(row[d] - m); row[d] = e; s += e; }
    const float inv = 1.f / s;
    for (int d = 0; d < CC; ++d) row[d] *= inv;
}

// ---------------------------------------------------------------------------
// Kernel 3: M[n] = conv_w @ softmax(G[n])  (128x128x128, trivial FLOPs)
// ---------------------------------------------------------------------------
__global__ __launch_bounds__(256) void mmat_kernel(const float* __restrict__ cw,
                                                   const float* __restrict__ Wt,
                                                   float* __restrict__ M) {
    const int n   = blockIdx.y;
    const int idx = blockIdx.x * 256 + threadIdx.x;   // 0..16383
    const int o   = idx >> 7;
    const int d   = idx & 127;
    const float* w = Wt + (size_t)n * CC * CC;
    float s = 0.f;
    for (int c = 0; c < CC; ++c) s += cw[o * CC + c] * w[c * CC + d];
    M[(size_t)n * CC * CC + idx] = s;
}

// ---------------------------------------------------------------------------
// Kernel 4: out[n][:, cb:cb+128] = M[n] @ X[n][:, cb:cb+128] + bias
// grid = (128 col blocks, 16 batches), 256 threads = 8 waves.
// X tile staged TRANSPOSED in LDS (B needs X columns); A-frags (rows of M)
// read directly from global fp32 (wave-private rows -> no duplication).
// ---------------------------------------------------------------------------
__global__ __launch_bounds__(256) void attn_out_kernel(const float* __restrict__ x,
                                                       const float* __restrict__ M,
                                                       const float* __restrict__ bias,
                                                       float* __restrict__ out) {
    // XT[l][c] bf16, row stride 136 shorts (272B): 16B-aligned, conflict-free.
    __shared__ __attribute__((aligned(16))) unsigned short ldsXT[CC * 136];

    const int n    = blockIdx.y;
    const int cb   = blockIdx.x * 128;
    const int t    = threadIdx.x;
    const int wave = t >> 5;
    const int lane = t & 31;
    const int half = lane >> 4;
    const int ln   = lane & 15;
    const float* xb = x + (size_t)n * CC * LL;
    const float* Mb = M + (size_t)n * CC * CC;

    // stage X[c][cb+l] -> ldsXT[l][c] as bf16 (coalesced reads, scattered b16 writes)
    for (int i = 0; i < 16; ++i) {
        int f  = t + 256 * i;            // float4 index 0..4095
        int c  = f >> 5;                 // 32 float4 per row
        int l4 = (f & 31) * 4;
        const float4 v = *(const float4*)(xb + (size_t)c * LL + cb + l4);
        ldsXT[(l4 + 0) * 136 + c] = f2bf(v.x);
        ldsXT[(l4 + 1) * 136 + c] = f2bf(v.y);
        ldsXT[(l4 + 2) * 136 + c] = f2bf(v.z);
        ldsXT[(l4 + 3) * 136 + c] = f2bf(v.w);
    }
    __syncthreads();

    v8f acc[8] = {};
    const int m = 16 * wave + ln;
    for (int ks = 0; ks < 4; ++ks) {      // K = 128 channels = 4 steps of 32
        v16bf a = frag_f32(Mb + (size_t)m * CC + ks * 32 + half * 8);
        for (int j = 0; j < 8; ++j) {
            const int l = 16 * j + ln;
            v16bf b = make_frag(&ldsXT[l * 136 + ks * 32 + half * 8],
                                &ldsXT[l * 136 + ks * 32 + 16 + half * 8]);
            acc[j] = __builtin_amdgcn_wmma_f32_16x16x32_bf16(
                         false, a, false, b, (short)0, acc[j], false, false);
        }
    }

    float bv[8];
    for (int v = 0; v < 8; ++v) bv[v] = bias[16 * wave + v + 8 * half];

    float* ob = out + (size_t)n * CC * LL;
    for (int j = 0; j < 8; ++j) {
        const int l = cb + 16 * j + ln;
        for (int v = 0; v < 8; ++v) {
            const int o = 16 * wave + v + 8 * half;
            ob[(size_t)o * LL + l] = acc[j][v] + bv[v];
        }
    }
}

// ---------------------------------------------------------------------------
extern "C" void kernel_launch(void* const* d_in, const int* in_sizes, int n_in,
                              void* d_out, int out_size, void* d_ws, size_t ws_size,
                              hipStream_t stream) {
    const float* x  = (const float*)d_in[0];   // (16,128,128,128) fp32
    const float* cw = (const float*)d_in[1];   // (128,128) fp32
    const float* cb = (const float*)d_in[2];   // (128,) fp32
    float* out = (float*)d_out;                // (16,128,128,128) fp32

    float* G = (float*)d_ws;                   // 16*128*128 fp32 (1 MB)
    float* M = G + (size_t)NB * CC * CC;       // 16*128*128 fp32 (1 MB)

    hipMemsetAsync(G, 0, (size_t)NB * CC * CC * sizeof(float), stream);

    gram_kernel<<<dim3(LL / KCHUNK, NB), 256, 0, stream>>>(x, G);     // 256 WGs
    softmax_kernel<<<NB, 128, 0, stream>>>(G);
    mmat_kernel<<<dim3(64, NB), 256, 0, stream>>>(cw, G, M);
    attn_out_kernel<<<dim3(LL / 128, NB), 256, 0, stream>>>(x, M, cb, out);
}
